// KfRNNCell_5669356831245
// MI455X (gfx1250) — matrix operand order repaired and checked by reading, not compiled, not run on hardware
//
#include <hip/hip_runtime.h>
#include <math.h>

// Problem constants (from reference): N_IN=1024, N=2048
#define N_IN  1024
#define NN    2048
#define HQ_LEN (N_IN + NN + 1)   // 3073
#define EPSF  1e-8f
// Signs drawn from a fixed RNG key in the reference; pinned here.
#define C1    1.0f
#define C2    1.0f

typedef __attribute__((ext_vector_type(2)))  float  v2f;
typedef __attribute__((ext_vector_type(8)))  float  v8f;
typedef __attribute__((ext_vector_type(8)))  __bf16 v8bf;
typedef __attribute__((ext_vector_type(16))) __bf16 v16bf;

// Workspace layout:
//  floats [0..2320):
//   [0] sum u_prev^2  [1] sum h_q^2  [2] sum dh_dz^2
//   [6] c1*p1  [7] c2*p2  [8] c1/(p1+eps)  [9] c2/(p2+eps)
//   [16..16+2048)      dh_dz
//   [2064..2064+256)   per-block ||H_q||^2 partials
//  bytes  [16384 ..): bf16 split operands (optional fast path, 32 MB)
#define WS_DH      16
#define WS_HQP     (16 + NN)
#define WS_BF_OFF  16384u
#define BF_ARR_BYTES ((size_t)NN * NN * sizeof(__bf16))   // 8 MB each

#define KSTEP   32      // K per stage step (one bf16 WMMA K)
#define NSTEP   (NN / KSTEP)
#define APITCH  40      // LDS row pitch in elements (80 B): banks 0,4,..,60 -> conflict-free

// ---------------------------------------------------------------------------
// Kernel 1: z = x@W_in + h_prev@W_h + bias ; h_next = tanh(z); dh = 1-h^2
// ---------------------------------------------------------------------------
__global__ __launch_bounds__(256)
void k_zvec(const float* __restrict__ x, const float* __restrict__ h_prev,
            const float* __restrict__ W_in, const float* __restrict__ W_h,
            const float* __restrict__ bias,
            float* __restrict__ h_out, float* __restrict__ ws) {
  const int j = blockIdx.x * 256 + threadIdx.x;   // 0..2047
  float z = bias[j];
#pragma unroll 4
  for (int i = 0; i < N_IN; ++i)
    z = fmaf(x[i], W_in[(size_t)i * NN + j], z);
#pragma unroll 4
  for (int i = 0; i < NN; ++i)
    z = fmaf(h_prev[i], W_h[(size_t)i * NN + j], z);
  const float h = tanhf(z);
  const float d = 1.0f - h * h;
  h_out[j] = h;
  ws[WS_DH + j] = d;
}

// ---------------------------------------------------------------------------
// Kernel 2: small norms (single block, deterministic LDS tree reduction)
// ---------------------------------------------------------------------------
__global__ __launch_bounds__(1024)
void k_small_norms(const float* __restrict__ x, const float* __restrict__ h_prev,
                   const float* __restrict__ u_prev, float* __restrict__ ws) {
  __shared__ float red[1024];
  const int t = threadIdx.x;
  float su = 0.f, sh = 0.f, sd = 0.f;
  for (int i = t; i < HQ_LEN; i += 1024) { float v = u_prev[i]; su = fmaf(v, v, su); }
  for (int i = t; i < N_IN;   i += 1024) { float v = x[i];      sh = fmaf(v, v, sh); }
  for (int i = t; i < NN;     i += 1024) {
    float v = h_prev[i];     sh = fmaf(v, v, sh);
    float d = ws[WS_DH + i]; sd = fmaf(d, d, sd);
  }
  if (t == 0) sh += 1.0f;   // trailing 1 in h_q
  float vals[3] = {su, sh, sd};
#pragma unroll
  for (int k = 0; k < 3; ++k) {
    red[t] = vals[k];
    __syncthreads();
    for (int s = 512; s > 0; s >>= 1) {
      if (t < s) red[t] += red[t + s];
      __syncthreads();
    }
    if (t == 0) ws[k] = red[0];
    __syncthreads();
  }
}

// ---------------------------------------------------------------------------
// Split pre-pass A: W_h (f32, row-major) -> Ahi/Alo (bf16, row-major).
// ---------------------------------------------------------------------------
__global__ __launch_bounds__(256)
void k_splitA(const float* __restrict__ Wh, __bf16* __restrict__ Ahi,
              __bf16* __restrict__ Alo) {
  const size_t i = ((size_t)blockIdx.x * 256 + threadIdx.x) * 2;
  const float v0 = Wh[i], v1 = Wh[i + 1];
  const __bf16 h0 = (__bf16)v0, h1 = (__bf16)v1;
  Ahi[i] = h0;  Ahi[i + 1] = h1;
  Alo[i] = (__bf16)(v0 - (float)h0);
  Alo[i + 1] = (__bf16)(v1 - (float)h1);
}

// ---------------------------------------------------------------------------
// Split pre-pass B: A_prev (f32, [K][N]) -> Bhi/Blo (bf16, TRANSPOSED [N][K])
// via LDS-tiled 64x64 transpose; both global read and write coalesced.
// ---------------------------------------------------------------------------
__global__ __launch_bounds__(256)
void k_splitBT(const float* __restrict__ Ap, __bf16* __restrict__ Bhi,
               __bf16* __restrict__ Blo) {
  __shared__ float tile[64][65];
  const int tx = threadIdx.x & 63, ty = threadIdx.x >> 6;   // ty: 0..3
  const int kb = blockIdx.y * 64, nb = blockIdx.x * 64;
#pragma unroll
  for (int r = ty; r < 64; r += 4)
    tile[r][tx] = Ap[(size_t)(kb + r) * NN + (nb + tx)];
  __syncthreads();
#pragma unroll
  for (int r = ty; r < 64; r += 4) {
    const size_t o = (size_t)(nb + r) * NN + (kb + tx);
    const float v = tile[tx][r];
    const __bf16 h = (__bf16)v;
    Bhi[o] = h;
    Blo[o] = (__bf16)(v - (float)h);
  }
}

// ---------------------------------------------------------------------------
// Fragment assembly: two contiguous b128 chunks -> one v16bf WMMA fragment.
// ---------------------------------------------------------------------------
__device__ __forceinline__ v16bf frag16(const __bf16* p0, const __bf16* p1) {
  const v8bf a = *reinterpret_cast<const v8bf*>(p0);
  const v8bf b = *reinterpret_cast<const v8bf*>(p1);
  return __builtin_shufflevector(a, b, 0, 1, 2, 3, 4, 5, 6, 7,
                                 8, 9, 10, 11, 12, 13, 14, 15);
}

// ---------------------------------------------------------------------------
// bf16x3 GEMM, LDS-staged + double-buffered:
//   H_q = diag(dh) * (W_h @ A_prev) via V_WMMA_F32_16X16X32_BF16.
// Block 256 = 8 waves, block tile 128x128, wave tile 64x32 (4x2 WMMA tiles).
// Per K-step (32): cooperative stage of A[128x32] hi/lo + B[128x32] hi/lo
// into LDS (each thread: 2 contiguous b128 per array, prefetched one step
// ahead); waves read fragments with conflict-free ds_load_b128 (pitch 40).
// All per-lane addresses are loop-invariant -> no per-iteration VALU address
// chains feeding WMMA-source registers (kills the 4-NOP hazard clusters).
// A-frag layout: lane-half h gets K h*16..+15.
// B-frag layout: lane-half h gets K h*16..+15 (transposed [N][K] source).
// ---------------------------------------------------------------------------
__global__ __launch_bounds__(256)
void k_wmma_gemm_bf16(const __bf16* __restrict__ Ahi, const __bf16* __restrict__ Alo,
                      const __bf16* __restrict__ Bhi, const __bf16* __restrict__ Blo,
                      const float* __restrict__ ws_ro, float* __restrict__ Hq,
                      float* __restrict__ ws) {
  __shared__ __bf16 sA[2][2][128 * APITCH];   // [buf][hi/lo]
  __shared__ __bf16 sB[2][2][128 * APITCH];
  __shared__ float wred[8];

  const int t = threadIdx.x;
  const int lane = t & 31, wave = t >> 5;
  const int lm = lane & 15;
  const int h8 = (lane >> 4) * 8;
  const int h16 = (lane >> 4) * 16;
  const float* __restrict__ dh = ws_ro + WS_DH;

  // ---- staging map: chunk pair c0=2t,2t+1 -> same row, 16 contiguous elems
  const int c0 = t * 2;
  const int srow = c0 >> 2;            // 0..127
  const int spos = (c0 & 3) * 8;       // 0 or 16
  const size_t gA = (size_t)(blockIdx.y * 128 + srow) * NN + spos;
  const size_t gB = (size_t)(blockIdx.x * 128 + srow) * NN + spos;
  const __bf16* pAhi = Ahi + gA;  const __bf16* pAlo = Alo + gA;
  const __bf16* pBhi = Bhi + gB;  const __bf16* pBlo = Blo + gB;
  const int lOff = srow * APITCH + spos;

  // ---- consume map: loop-invariant LDS fragment offsets
  const int rowBase = blockIdx.y * 128 + (wave >> 2) * 64;
  const int colBase = blockIdx.x * 128 + (wave & 3) * 32;
  int laA[4], laB[2];
#pragma unroll
  for (int mt = 0; mt < 4; ++mt)
    laA[mt] = ((wave >> 2) * 64 + mt * 16 + lm) * APITCH + h16;
#pragma unroll
  for (int nt = 0; nt < 2; ++nt)
    laB[nt] = ((wave & 3) * 32 + nt * 16 + lm) * APITCH + h16;

  v8f acc[4][2] = {};
  v8bf rA[2][2], rB[2][2];   // prefetch regs [hi/lo][chunk]

#define FETCH(K0)                                                        \
  do {                                                                   \
    rA[0][0] = *(const v8bf*)(pAhi + (K0));                              \
    rA[0][1] = *(const v8bf*)(pAhi + (K0) + 8);                          \
    rA[1][0] = *(const v8bf*)(pAlo + (K0));                              \
    rA[1][1] = *(const v8bf*)(pAlo + (K0) + 8);                          \
    rB[0][0] = *(const v8bf*)(pBhi + (K0));                              \
    rB[0][1] = *(const v8bf*)(pBhi + (K0) + 8);                          \
    rB[1][0] = *(const v8bf*)(pBlo + (K0));                              \
    rB[1][1] = *(const v8bf*)(pBlo + (K0) + 8);                          \
  } while (0)

#define STORE(B)                                                         \
  do {                                                                   \
    *(v8bf*)&sA[B][0][lOff]     = rA[0][0];                              \
    *(v8bf*)&sA[B][0][lOff + 8] = rA[0][1];                              \
    *(v8bf*)&sA[B][1][lOff]     = rA[1][0];                              \
    *(v8bf*)&sA[B][1][lOff + 8] = rA[1][1];                              \
    *(v8bf*)&sB[B][0][lOff]     = rB[0][0];                              \
    *(v8bf*)&sB[B][0][lOff + 8] = rB[0][1];                              \
    *(v8bf*)&sB[B][1][lOff]     = rB[1][0];                              \
    *(v8bf*)&sB[B][1][lOff + 8] = rB[1][1];                              \
  } while (0)

  FETCH(0);
  STORE(0);
  __syncthreads();

  for (int s = 0; s < NSTEP; ++s) {
    const int b = s & 1;
    if (s + 1 < NSTEP) FETCH((s + 1) * KSTEP);

    v16bf bhiF[2], bloF[2];
#pragma unroll
    for (int nt = 0; nt < 2; ++nt) {
      bhiF[nt] = frag16(&sB[b][0][laB[nt]], &sB[b][0][laB[nt] + 8]);
      bloF[nt] = frag16(&sB[b][1][laB[nt]], &sB[b][1][laB[nt] + 8]);
    }
#pragma unroll
    for (int mt = 0; mt < 4; ++mt) {
      const v16bf ahiF = frag16(&sA[b][0][laA[mt]], &sA[b][0][laA[mt] + 8]);
      const v16bf aloF = frag16(&sA[b][1][laA[mt]], &sA[b][1][laA[mt] + 8]);
#pragma unroll
      for (int nt = 0; nt < 2; ++nt) {
        acc[mt][nt] = __builtin_amdgcn_wmma_f32_16x16x32_bf16(
            false, ahiF, false, bhiF[nt], (short)0, acc[mt][nt], false, false);
        acc[mt][nt] = __builtin_amdgcn_wmma_f32_16x16x32_bf16(
            false, ahiF, false, bloF[nt], (short)0, acc[mt][nt], false, false);
        acc[mt][nt] = __builtin_amdgcn_wmma_f32_16x16x32_bf16(
            false, aloF, false, bhiF[nt], (short)0, acc[mt][nt], false, false);
      }
    }
    if (s + 1 < NSTEP) {
      STORE(b ^ 1);
      __syncthreads();
    }
  }
#undef FETCH
#undef STORE

  // Epilogue: row scale by dh, store H_q, deterministic per-block sum-sq.
  float lsq = 0.f;
#pragma unroll
  for (int mt = 0; mt < 4; ++mt) {
#pragma unroll
    for (int nt = 0; nt < 2; ++nt) {
      const int c = colBase + nt * 16 + lm;
#pragma unroll
      for (int i = 0; i < 8; ++i) {
        const int r = rowBase + mt * 16 + i + (lane >> 4) * 8;  // C/D layout
        float v = acc[mt][nt][i] * dh[r];
        Hq[(size_t)r * NN + c] = v;
        lsq = fmaf(v, v, lsq);
      }
    }
  }
  for (int m = 16; m > 0; m >>= 1) lsq += __shfl_xor(lsq, m, 32);
  if (lane == 0) wred[wave] = lsq;
  __syncthreads();
  if (threadIdx.x == 0) {
    float sum = 0.f;
#pragma unroll
    for (int w = 0; w < 8; ++w) sum += wred[w];
    ws[WS_HQP + blockIdx.y * gridDim.x + blockIdx.x] = sum;
  }
}

// ---------------------------------------------------------------------------
// Fallback f32 GEMM (V_WMMA_F32_16X16X4_F32) if ws can't hold split buffers.
// ---------------------------------------------------------------------------
__global__ __launch_bounds__(256)
void k_wmma_gemm_f32(const float* __restrict__ Wh, const float* __restrict__ Ap,
                     const float* __restrict__ ws_ro, float* __restrict__ Hq,
                     float* __restrict__ ws) {
  __shared__ float wred[8];
  const int lane = threadIdx.x & 31;
  const int wave = threadIdx.x >> 5;
  const int rowBase = blockIdx.y * 128 + (wave >> 2) * 64;
  const int colBase = blockIdx.x * 128 + (wave & 3) * 32;
  const int lm = lane & 15;
  const int kh = (lane >> 4) * 2;
  const float* __restrict__ dh = ws_ro + WS_DH;

  v8f acc[4][2] = {};
  for (int k0 = 0; k0 < NN; k0 += 4) {
    v2f a[4];
#pragma unroll
    for (int mt = 0; mt < 4; ++mt) {
      const int r = rowBase + mt * 16 + lm;
      a[mt] = *reinterpret_cast<const v2f*>(Wh + (size_t)r * NN + (k0 + kh));
    }
    v2f b[2];
#pragma unroll
    for (int nt = 0; nt < 2; ++nt) {
      const int c = colBase + nt * 16 + lm;
      b[nt].x = Ap[(size_t)(k0 + kh) * NN + c];
      b[nt].y = Ap[(size_t)(k0 + kh + 1) * NN + c];
    }
#pragma unroll
    for (int mt = 0; mt < 4; ++mt)
#pragma unroll
      for (int nt = 0; nt < 2; ++nt)
        acc[mt][nt] = __builtin_amdgcn_wmma_f32_16x16x4_f32(
            false, a[mt], false, b[nt], (short)0, acc[mt][nt], false, false);
  }

  float lsq = 0.f;
#pragma unroll
  for (int mt = 0; mt < 4; ++mt) {
#pragma unroll
    for (int nt = 0; nt < 2; ++nt) {
      const int c = colBase + nt * 16 + lm;
#pragma unroll
      for (int i = 0; i < 8; ++i) {
        const int r = rowBase + mt * 16 + i + (lane >> 4) * 8;
        float v = acc[mt][nt][i] * dh[r];
        Hq[(size_t)r * NN + c] = v;
        lsq = fmaf(v, v, lsq);
      }
    }
  }
  for (int m = 16; m > 0; m >>= 1) lsq += __shfl_xor(lsq, m, 32);
  if (lane == 0) wred[wave] = lsq;
  __syncthreads();
  if (threadIdx.x == 0) {
    float s = 0.f;
#pragma unroll
    for (int w = 0; w < 8; ++w) s += wred[w];
    ws[WS_HQP + blockIdx.y * gridDim.x + blockIdx.x] = s;
  }
}

// ---------------------------------------------------------------------------
__global__ __launch_bounds__(256)
void k_scalars(float* __restrict__ ws) {
  __shared__ float red[256];
  const int t = threadIdx.x;
  red[t] = ws[WS_HQP + t];
  __syncthreads();
  for (int s = 128; s > 0; s >>= 1) {
    if (t < s) red[t] += red[t + s];
    __syncthreads();
  }
  if (t == 0) {
    const float nH  = sqrtf(red[0]);
    const float nu  = sqrtf(ws[0]);
    const float nhq = sqrtf(ws[1]);
    const float nd  = sqrtf(ws[2]);
    const float p1 = sqrtf(nH / (EPSF + nu));
    const float p2 = sqrtf(nd / (EPSF + nhq));
    ws[6] = C1 * p1;
    ws[7] = C2 * p2;
    ws[8] = C1 / (p1 + EPSF);
    ws[9] = C2 / (p2 + EPSF);
  }
}

__global__ __launch_bounds__(256)
void k_unext(const float* __restrict__ x, const float* __restrict__ h_prev,
             const float* __restrict__ u_prev, const float* __restrict__ ws,
             float* __restrict__ u_out) {
  const int i = blockIdx.x * 256 + threadIdx.x;
  if (i >= HQ_LEN) return;
  const float hq = (i < N_IN) ? x[i] : ((i < N_IN + NN) ? h_prev[i - N_IN] : 1.0f);
  u_out[i] = ws[6] * u_prev[i] + ws[7] * hq;
}

__global__ __launch_bounds__(256)
void k_anext(float* __restrict__ A, const float* __restrict__ ws) {
  const int idx = blockIdx.x * 256 + threadIdx.x;   // < 2048*2048
  const int r = idx >> 11, c = idx & (NN - 1);
  float v = A[idx] * ws[8];
  if (r == c) v = fmaf(ws[9], ws[WS_DH + r], v);
  A[idx] = v;
}

// ---------------------------------------------------------------------------
extern "C" void kernel_launch(void* const* d_in, const int* in_sizes, int n_in,
                              void* d_out, int out_size, void* d_ws, size_t ws_size,
                              hipStream_t stream) {
  const float* x      = (const float*)d_in[0];
  const float* h_prev = (const float*)d_in[1];
  const float* W_in   = (const float*)d_in[2];
  const float* W_h    = (const float*)d_in[3];
  const float* bias   = (const float*)d_in[4];
  const float* u_prev = (const float*)d_in[5];
  const float* A_prev = (const float*)d_in[6];
  float* out = (float*)d_out;
  float* ws  = (float*)d_ws;

  float* h_out = out;                  // [2048]
  float* u_out = out + NN;             // [3073]
  float* A_out = out + NN + HQ_LEN;    // [2048*2048], holds H_q then A_next

  k_zvec<<<dim3(NN / 256), dim3(256), 0, stream>>>(x, h_prev, W_in, W_h, bias,
                                                   h_out, ws);
  k_small_norms<<<dim3(1), dim3(1024), 0, stream>>>(x, h_prev, u_prev, ws);

  const size_t need_bf16 = (size_t)WS_BF_OFF + 4 * BF_ARR_BYTES;  // ~32 MB
  if (ws_size >= need_bf16) {
    // Fast path: split-bf16 GEMM (~2.7x f32-WMMA rate, ~3e-5 rel. error).
    __bf16* Ahi = (__bf16*)((char*)d_ws + WS_BF_OFF);
    __bf16* Alo = (__bf16*)((char*)d_ws + WS_BF_OFF + 1 * BF_ARR_BYTES);
    __bf16* Bhi = (__bf16*)((char*)d_ws + WS_BF_OFF + 2 * BF_ARR_BYTES);
    __bf16* Blo = (__bf16*)((char*)d_ws + WS_BF_OFF + 3 * BF_ARR_BYTES);
    k_splitA<<<dim3((NN * NN / 2) / 256), dim3(256), 0, stream>>>(W_h, Ahi, Alo);
    k_splitBT<<<dim3(32, 32), dim3(256), 0, stream>>>(A_prev, Bhi, Blo);
    k_wmma_gemm_bf16<<<dim3(16, 16), dim3(256), 0, stream>>>(Ahi, Alo, Bhi, Blo,
                                                             ws, A_out, ws);
  } else {
    // Fallback: full-f32 WMMA GEMM, no extra workspace.
    k_wmma_gemm_f32<<<dim3(16, 16), dim3(256), 0, stream>>>(W_h, A_prev, ws,
                                                            A_out, ws);
  }

  k_scalars<<<dim3(1), dim3(256), 0, stream>>>(ws);
  k_unext<<<dim3((HQ_LEN + 255) / 256), dim3(256), 0, stream>>>(x, h_prev,
                                                                u_prev, ws, u_out);
  k_anext<<<dim3((NN * NN) / 256), dim3(256), 0, stream>>>(A_out, ws);
}
